// GNN_54357106098458
// MI455X (gfx1250) — compile-verified, matrix operand order checked by reading, not run on hardware
//
#include <hip/hip_runtime.h>
#include <hip/hip_bf16.h>

// ---------------------------------------------------------------------------
// GIN + virtual node GNN for MI455X (gfx1250, wave32, WMMA).
// GEMMs use v_wmma_f32_16x16x32_bf16 (bf16 inputs, f32 accumulate).
// Scatter/segment sums use global f32 atomics; BN folds stats into
// per-column scale/shift then applies as vectorized FMA.
// K loop is split into a mask-free main body + masked tail so the steady
// state is pure b128 global loads -> pk-bf16 converts -> b64 (A) / b16 (W^T) LDS stores.
// ---------------------------------------------------------------------------

#define N_ATOM   262144
#define E_EDGE   524288
#define G_GRAPH  8192
#define DIM      300
#define DIM4     75     // DIM/4
#define DIM2     600
#define LAYERS   5
#define NUM_TASK 128
#define FP_DIM   2048
#define HEAD_H   1200   // 4*DIM
#define HEAD_K   2348   // DIM + FP_DIM

typedef __attribute__((ext_vector_type(16))) __bf16 v16bf;
typedef __attribute__((ext_vector_type(8)))  float  v8f;

union Frag  { unsigned int u[8]; v16bf v; };
union Pack4 { __bf16 h[4]; unsigned long long u64; };

// -------------------------------- WMMA GEMM --------------------------------
// C[M,Ncols] = A[M,K] @ W[K,Ncols] (+ bias | += existing C)
// REQUIREMENTS: M % 128 == 0, K % 4 == 0, Ncols % 4 == 0, pointers 16B aligned.
// Block: 256 threads (8 waves). Tile: 128 rows x 64 cols, K step 32.
#define TM 128
#define TN 64
#define TK 32

static __device__ __forceinline__ void frag_mma(
    const __bf16 (*sA)[TK + 8], const __bf16 (*sB)[TK + 8],
    v8f acc[4], int waveRow, int half, int l16)
{
    // A fragment per ISA 16-bit 16x32 layout:
    // lane L: M = L%16; vgpr v holds K = (v/4)*16 + (L/16)*8 + (v%4)*2 (+0/1)
    Frag fa;
    const __bf16* ap = &sA[waveRow + l16][0];
    #pragma unroll
    for (int v = 0; v < 8; ++v) {
        int kb = ((v >> 2) << 4) + (half << 3) + ((v & 3) << 1);
        fa.u[v] = *(const unsigned int*)(ap + kb);   // 2 bf16 (K, K+1)
    }
    #pragma unroll
    for (int c = 0; c < 4; ++c) {
        Frag fb;
        const __bf16* bp = &sB[c * 16 + l16][0];
        #pragma unroll
        for (int v = 0; v < 8; ++v) {
            int kb = ((v >> 2) << 4) + (half << 3) + ((v & 3) << 1);
            fb.u[v] = *(const unsigned int*)(bp + kb);
        }
        acc[c] = __builtin_amdgcn_wmma_f32_16x16x32_bf16(
            false, fa.v, false, fb.v, (short)0, acc[c], false, false);
    }
}

__launch_bounds__(256)
__global__ void gemm_bf16_wmma(const float* __restrict__ A, int lda,
                               const float* __restrict__ W, int ldw,
                               const float* __restrict__ bias,
                               float* __restrict__ C, int ldc,
                               int M, int K, int Ncols, int accum)
{
    __shared__ __bf16 sA[TM][TK + 8];      // row-major A tile (bf16)
    __shared__ __bf16 sB[TN][TK + 8];      // W tile stored transposed [n][k]

    const int tid  = threadIdx.x;
    const int wave = tid >> 5;
    const int lane = tid & 31;
    const int half = lane >> 4;
    const int l16  = lane & 15;

    const int rowBase = blockIdx.x * TM;
    const int colBase = blockIdx.y * TN;
    const int waveRow = wave * 16;

    // A staging: 2 threads per row, 16 contiguous k (4 float4) each
    const int ar  = tid >> 1;              // 0..127
    const int akk = (tid & 1) << 4;        // 0 or 16
    const float* Arow = A + (size_t)(rowBase + ar) * lda + akk;

    // W staging: 16 threads per k-row, 4 contiguous n each (x2 k-iters)
    const int wkk = tid >> 4;              // 0..15 (+16 on second iter)
    const int wnc = (tid & 15) << 2;       // 0..60
    const int gc       = colBase + wnc;
    const bool colOK   = gc < Ncols;
    const int gc_c     = colOK ? gc : (Ncols - 4);   // Ncols % 4 == 0
    const float* Wbase = W + (size_t)wkk * ldw + gc_c;

    v8f acc[4];
    acc[0] = v8f{}; acc[1] = v8f{}; acc[2] = v8f{}; acc[3] = v8f{};

    const int kFull = K & ~(TK - 1);       // main-loop span (no masking needed)

    // ------------------------- mask-free main loop -------------------------
    for (int k0 = 0; k0 < kFull; k0 += TK) {
        float4 va[4];
        #pragma unroll
        for (int j = 0; j < 4; ++j)
            va[j] = *(const float4*)(Arow + k0 + j * 4);
        float4 wv[2];
        #pragma unroll
        for (int it = 0; it < 2; ++it)
            wv[it] = *(const float4*)(Wbase + (size_t)(k0 + it * 16) * ldw);

        if (k0 + TK < K)
            __builtin_prefetch(Arow + k0 + TK, 0, 0);

        #pragma unroll
        for (int j = 0; j < 4; ++j) {
            Pack4 p;
            p.h[0] = (__bf16)va[j].x; p.h[1] = (__bf16)va[j].y;
            p.h[2] = (__bf16)va[j].z; p.h[3] = (__bf16)va[j].w;
            *(unsigned long long*)(&sA[ar][akk + j * 4]) = p.u64;
        }
        #pragma unroll
        for (int it = 0; it < 2; ++it) {
            float4 v = wv[it];
            if (!colOK) v = make_float4(0.f, 0.f, 0.f, 0.f);
            int kk = wkk + it * 16;
            sB[wnc + 0][kk] = (__bf16)v.x;
            sB[wnc + 1][kk] = (__bf16)v.y;
            sB[wnc + 2][kk] = (__bf16)v.z;
            sB[wnc + 3][kk] = (__bf16)v.w;
        }
        __syncthreads();
        frag_mma(sA, sB, acc, waveRow, half, l16);
        __syncthreads();
    }

    // ----------------------------- masked tail -----------------------------
    if (kFull < K) {
        const int k0 = kFull;
        #pragma unroll
        for (int j = 0; j < 4; ++j) {
            int gk   = k0 + akk + j * 4;
            int gk_c = (gk < K) ? gk : (K - 4);          // clamp, K % 4 == 0
            float4 v = *(const float4*)(A + (size_t)(rowBase + ar) * lda + gk_c);
            if (gk >= K) v = make_float4(0.f, 0.f, 0.f, 0.f);
            Pack4 p;
            p.h[0] = (__bf16)v.x; p.h[1] = (__bf16)v.y;
            p.h[2] = (__bf16)v.z; p.h[3] = (__bf16)v.w;
            *(unsigned long long*)(&sA[ar][akk + j * 4]) = p.u64;
        }
        #pragma unroll
        for (int it = 0; it < 2; ++it) {
            int kk   = wkk + it * 16;
            int gk   = k0 + kk;
            int gk_c = (gk < K) ? gk : (K - 1);
            float4 v = *(const float4*)(W + (size_t)gk_c * ldw + gc_c);
            if (gk >= K || !colOK) v = make_float4(0.f, 0.f, 0.f, 0.f);
            sB[wnc + 0][kk] = (__bf16)v.x;
            sB[wnc + 1][kk] = (__bf16)v.y;
            sB[wnc + 2][kk] = (__bf16)v.z;
            sB[wnc + 3][kk] = (__bf16)v.w;
        }
        __syncthreads();
        frag_mma(sA, sB, acc, waveRow, half, l16);
        __syncthreads();
    }

    // ---- store: C/D layout — VGPR r holds M = r + 8*(lane/16), N = lane%16 ----
    #pragma unroll
    for (int c = 0; c < 4; ++c) {
        int col = colBase + c * 16 + l16;
        if (col < Ncols) {
            float badd = bias ? bias[col] : 0.0f;
            #pragma unroll
            for (int r = 0; r < 8; ++r) {
                int row = rowBase + waveRow + half * 8 + r;   // row < M guaranteed
                size_t off = (size_t)row * ldc + col;
                float v = acc[c][r];
                if (accum) C[off] += v;
                else       C[off] = v + badd;
            }
        }
    }
}

// ----------------------------- elementwise kernels -------------------------

__global__ void k_zero4(float4* p, size_t n4) {
    size_t i = (size_t)blockIdx.x * blockDim.x + threadIdx.x;
    if (i < n4) p[i] = make_float4(0.f, 0.f, 0.f, 0.f);
}

__global__ void k_copy4(float4* dst, const float4* src, size_t n4) {
    size_t i = (size_t)blockIdx.x * blockDim.x + threadIdx.x;
    if (i < n4) dst[i] = src[i];
}

// h[i,:] = atom_emb[x_atom[i],:]   grid(N), block 96 (75 float4 per row)
__global__ void k_embed(const int* __restrict__ x_atom,
                        const float* __restrict__ atom_emb,
                        float* __restrict__ h) {
    int i = blockIdx.x;
    int d = threadIdx.x;
    if (d >= DIM4) return;
    ((float4*)h)[(size_t)i * DIM4 + d] =
        ((const float4*)atom_emb)[(size_t)x_atom[i] * DIM4 + d];
}

// hl[i,:] = h[i,:] + vn[batch[i],:]
__global__ void k_add_vn(const float* __restrict__ h,
                         const float* __restrict__ vn,
                         const int* __restrict__ batch,
                         float* __restrict__ hl) {
    int i = blockIdx.x;
    int d = threadIdx.x;
    if (d >= DIM4) return;
    float4 a = ((const float4*)h)[(size_t)i * DIM4 + d];
    float4 b = ((const float4*)vn)[(size_t)batch[i] * DIM4 + d];
    ((float4*)hl)[(size_t)i * DIM4 + d] =
        make_float4(a.x + b.x, a.y + b.y, a.z + b.z, a.w + b.w);
}

// aggr[dst[e],:] += relu(hl[src[e],:] + edge_emb_l[edge_attr[e],:])
__global__ void k_edge_scatter(const float* __restrict__ hl,
                               const int* __restrict__ src,
                               const int* __restrict__ dst,
                               const int* __restrict__ eattr,
                               const float* __restrict__ eemb_l,
                               float* __restrict__ aggr) {
    int e = blockIdx.x;
    int d = threadIdx.x;
    if (d >= DIM4) return;
    int s = src[e], t = dst[e], a = eattr[e];
    float4 hv = ((const float4*)hl)[(size_t)s * DIM4 + d];
    float4 ev = ((const float4*)eemb_l)[(size_t)a * DIM4 + d];
    float m0 = hv.x + ev.x; m0 = m0 > 0.f ? m0 : 0.f;
    float m1 = hv.y + ev.y; m1 = m1 > 0.f ? m1 : 0.f;
    float m2 = hv.z + ev.z; m2 = m2 > 0.f ? m2 : 0.f;
    float m3 = hv.w + ev.w; m3 = m3 > 0.f ? m3 : 0.f;
    float* ap = aggr + (size_t)t * DIM + d * 4;
    atomicAdd(ap + 0, m0);
    atomicAdd(ap + 1, m1);
    atomicAdd(ap + 2, m2);
    atomicAdd(ap + 3, m3);
}

// z = (1+eps[l])*hl + aggr  (in-place into aggr), float4
__global__ void k_zcombine(float4* __restrict__ aggr,
                           const float4* __restrict__ hl,
                           const float* __restrict__ eps, int l, size_t n4) {
    size_t i = (size_t)blockIdx.x * blockDim.x + threadIdx.x;
    if (i >= n4) return;
    float a = 1.0f + eps[l];
    float4 h = hl[i], g = aggr[i];
    aggr[i] = make_float4(a * h.x + g.x, a * h.y + g.y,
                          a * h.z + g.z, a * h.w + g.w);
}

// column sum & sumsq, block-partial then atomic merge. grid(ceil(M/256), ceil(C/256))
__global__ void k_bn_stats(const float* __restrict__ x, float* __restrict__ stats,
                           int M, int C) {
    int c = blockIdx.y * blockDim.x + threadIdx.x;
    if (c >= C) return;
    int r0 = blockIdx.x * 256;
    int r1 = r0 + 256; if (r1 > M) r1 = M;
    float s = 0.0f, q = 0.0f;
    for (int r = r0; r < r1; ++r) {
        float v = x[(size_t)r * C + c];
        s += v; q += v * v;
    }
    atomicAdd(&stats[c], s);
    atomicAdd(&stats[C + c], q);
}

// fold stats into per-column scale/shift: y = x*scale + shift
__global__ void k_bn_finalize(float* __restrict__ stats,
                              const float* __restrict__ gamma,
                              const float* __restrict__ beta, int M, int C) {
    int c = blockIdx.x * blockDim.x + threadIdx.x;
    if (c >= C) return;
    float invM = 1.0f / (float)M;
    float mu  = stats[c] * invM;
    float var = stats[C + c] * invM - mu * mu;
    float sc  = gamma[c] * rsqrtf(var + 1e-5f);
    stats[2 * C + c] = sc;
    stats[3 * C + c] = beta[c] - mu * sc;
}

// y = x*scale + shift (+relu), float4, grid(M, ceil(C/4/128)), block 128
__global__ void k_bn_apply4(float* __restrict__ x, const float* __restrict__ stats,
                            int C, int relu) {
    int row = blockIdx.x;
    int C4  = C >> 2;
    int c4  = blockIdx.y * blockDim.x + threadIdx.x;
    if (c4 >= C4) return;
    float4 sc = ((const float4*)(stats + 2 * C))[c4];
    float4 sh = ((const float4*)(stats + 3 * C))[c4];
    float4* xp = (float4*)x + (size_t)row * C4 + c4;
    float4 v = *xp;
    v.x = fmaf(v.x, sc.x, sh.x);
    v.y = fmaf(v.y, sc.y, sh.y);
    v.z = fmaf(v.z, sc.z, sh.z);
    v.w = fmaf(v.w, sc.w, sh.w);
    if (relu) {
        v.x = v.x > 0.f ? v.x : 0.f;
        v.y = v.y > 0.f ? v.y : 0.f;
        v.z = v.z > 0.f ? v.z : 0.f;
        v.w = v.w > 0.f ? v.w : 0.f;
    }
    *xp = v;
}

// out[seg[i],:] += x[i,:]    grid(N), block 96
__global__ void k_seg_sum(const float* __restrict__ x,
                          const int* __restrict__ seg,
                          float* __restrict__ out) {
    int i = blockIdx.x;
    int d = threadIdx.x;
    if (d >= DIM4) return;
    float4 v = ((const float4*)x)[(size_t)i * DIM4 + d];
    float* op = out + (size_t)seg[i] * DIM + d * 4;
    atomicAdd(op + 0, v.x);
    atomicAdd(op + 1, v.y);
    atomicAdd(op + 2, v.z);
    atomicAdd(op + 3, v.w);
}

// tanh-approx GELU (jax.nn.gelu default), float4
__global__ void k_gelu4(float4* __restrict__ x, size_t n4) {
    size_t i = (size_t)blockIdx.x * blockDim.x + threadIdx.x;
    if (i >= n4) return;
    float4 v = x[i];
    float c0 = 0.7978845608028654f, c1 = 0.044715f;
    v.x = 0.5f * v.x * (1.0f + tanhf(c0 * (v.x + c1 * v.x * v.x * v.x)));
    v.y = 0.5f * v.y * (1.0f + tanhf(c0 * (v.y + c1 * v.y * v.y * v.y)));
    v.z = 0.5f * v.z * (1.0f + tanhf(c0 * (v.z + c1 * v.z * v.z * v.z)));
    v.w = 0.5f * v.w * (1.0f + tanhf(c0 * (v.w + c1 * v.w * v.w * v.w)));
    x[i] = v;
}

// --------------------------------- launch ----------------------------------

static inline size_t cdiv(size_t a, size_t b) { return (a + b - 1) / b; }

extern "C" void kernel_launch(void* const* d_in, const int* in_sizes, int n_in,
                              void* d_out, int out_size, void* d_ws, size_t ws_size,
                              hipStream_t stream) {
    (void)in_sizes; (void)n_in; (void)out_size; (void)ws_size;

    // ---- inputs (setup_inputs order) ----
    const int*   x_atom   = (const int*)  d_in[0];
    const int*   src      = (const int*)  d_in[1];
    const int*   dst      = (const int*)  d_in[2];
    const int*   eattr    = (const int*)  d_in[3];
    const int*   batch    = (const int*)  d_in[4];
    const float* fp       = (const float*)d_in[5];
    const float* atom_emb = (const float*)d_in[6];
    const float* edge_emb = (const float*)d_in[7];   // [L,5,D]
    const float* eps      = (const float*)d_in[8];
    const float* W1       = (const float*)d_in[9];   // [L,D,2D]
    const float* b1       = (const float*)d_in[10];
    const float* g1       = (const float*)d_in[11];
    const float* be1      = (const float*)d_in[12];
    const float* W2       = (const float*)d_in[13];  // [L,2D,D]
    const float* b2       = (const float*)d_in[14];
    const float* bng      = (const float*)d_in[15];
    const float* bnb      = (const float*)d_in[16];
    const float* vW1      = (const float*)d_in[17];
    const float* vb1      = (const float*)d_in[18];
    const float* vg1      = (const float*)d_in[19];
    const float* vbe1     = (const float*)d_in[20];
    const float* vW2      = (const float*)d_in[21];
    const float* vb2      = (const float*)d_in[22];
    const float* vg2      = (const float*)d_in[23];
    const float* vbe2     = (const float*)d_in[24];
    const float* pW1      = (const float*)d_in[25];  // [2348,1200]
    const float* pb1      = (const float*)d_in[26];
    const float* pW2      = (const float*)d_in[27];  // [1200,128]
    const float* pb2      = (const float*)d_in[28];
    float* out = (float*)d_out;

    // ---- workspace carve-up (floats) ----
    float* ws = (float*)d_ws;
    size_t off = 0;
    float* h     = ws + off; off += (size_t)N_ATOM * DIM;     // 78.6M
    float* hl    = ws + off; off += (size_t)N_ATOM * DIM;
    float* zbuf  = ws + off; off += (size_t)N_ATOM * DIM;     // aggr, then z in-place
    float* t1    = ws + off; off += (size_t)N_ATOM * DIM2;    // [N,600]
    float* vn    = ws + off; off += (size_t)G_GRAPH * DIM;
    float* vt    = ws + off; off += (size_t)G_GRAPH * DIM;
    float* vtmp  = ws + off; off += (size_t)G_GRAPH * DIM2;
    float* hg    = ws + off; off += (size_t)G_GRAPH * DIM;
    float* th    = ws + off; off += (size_t)G_GRAPH * HEAD_H;
    float* stats = ws + off; off += 4 * DIM2;                 // sum|sumsq|scale|shift

    const dim3 bk96(96);
    const dim3 gridN((unsigned)N_ATOM);
    const dim3 gridE((unsigned)E_EDGE);

    auto zero = [&](float* p, size_t n) {   // n % 4 == 0 for all call sites
        size_t n4 = n / 4;
        k_zero4<<<dim3((unsigned)cdiv(n4, 256)), dim3(256), 0, stream>>>(
            (float4*)p, n4);
    };
    auto gemm = [&](const float* A, int lda, const float* W, int ldw,
                    const float* bias, float* C, int ldc,
                    int M, int K, int Nc, int accum) {
        dim3 g((unsigned)cdiv(M, TM), (unsigned)cdiv(Nc, TN));
        gemm_bf16_wmma<<<g, dim3(256), 0, stream>>>(A, lda, W, ldw, bias, C, ldc,
                                                    M, K, Nc, accum);
    };
    auto bn = [&](float* x, const float* gamma, const float* beta,
                  int M, int C, int relu) {
        zero(stats, 2 * (size_t)C);
        dim3 gs((unsigned)cdiv(M, 256), (unsigned)cdiv(C, 256));
        k_bn_stats<<<gs, dim3(256), 0, stream>>>(x, stats, M, C);
        k_bn_finalize<<<dim3((unsigned)cdiv(C, 256)), dim3(256), 0, stream>>>(
            stats, gamma, beta, M, C);
        dim3 ga((unsigned)M, (unsigned)cdiv(C / 4, 128));
        k_bn_apply4<<<ga, dim3(128), 0, stream>>>(x, stats, C, relu);
    };

    // h = atom_emb[x_atom];  vn = 0
    k_embed<<<gridN, bk96, 0, stream>>>(x_atom, atom_emb, h);
    zero(vn, (size_t)G_GRAPH * DIM);

    for (int l = 0; l < LAYERS; ++l) {
        // hl = h + vn[batch]
        k_add_vn<<<gridN, bk96, 0, stream>>>(h, vn, batch, hl);

        // aggr = scatter_add(relu(hl[src] + edge_emb[l][edge_attr]), dst)
        zero(zbuf, (size_t)N_ATOM * DIM);
        k_edge_scatter<<<gridE, bk96, 0, stream>>>(
            hl, src, dst, eattr, edge_emb + (size_t)l * 5 * DIM, zbuf);

        // z = (1+eps[l])*hl + aggr   (in-place in zbuf)
        {
            size_t n4 = (size_t)N_ATOM * DIM / 4;
            k_zcombine<<<dim3((unsigned)cdiv(n4, 256)), dim3(256), 0, stream>>>(
                (float4*)zbuf, (const float4*)hl, eps, l, n4);
        }

        // t1 = z @ W1[l] + b1[l];  BN+relu
        gemm(zbuf, DIM, W1 + (size_t)l * DIM * DIM2, DIM2,
             b1 + (size_t)l * DIM2, t1, DIM2, N_ATOM, DIM, DIM2, 0);
        bn(t1, g1 + (size_t)l * DIM2, be1 + (size_t)l * DIM2, N_ATOM, DIM2, 1);

        // h = BN(t1 @ W2[l] + b2[l])  (+relu unless last layer)
        gemm(t1, DIM2, W2 + (size_t)l * DIM2 * DIM, DIM,
             b2 + (size_t)l * DIM, h, DIM, N_ATOM, DIM2, DIM, 0);
        bn(h, bng + (size_t)l * DIM, bnb + (size_t)l * DIM,
           N_ATOM, DIM, (l < LAYERS - 1) ? 1 : 0);

        // virtual node update
        if (l < LAYERS - 1) {
            size_t nv4 = (size_t)G_GRAPH * DIM / 4;
            k_copy4<<<dim3((unsigned)cdiv(nv4, 256)), dim3(256), 0, stream>>>(
                (float4*)vt, (const float4*)vn, nv4);
            k_seg_sum<<<gridN, bk96, 0, stream>>>(hl, batch, vt);

            gemm(vt, DIM, vW1 + (size_t)l * DIM * DIM2, DIM2,
                 vb1 + (size_t)l * DIM2, vtmp, DIM2, G_GRAPH, DIM, DIM2, 0);
            bn(vtmp, vg1 + (size_t)l * DIM2, vbe1 + (size_t)l * DIM2,
               G_GRAPH, DIM2, 1);

            gemm(vtmp, DIM2, vW2 + (size_t)l * DIM2 * DIM, DIM,
                 vb2 + (size_t)l * DIM, vn, DIM, G_GRAPH, DIM2, DIM, 0);
            bn(vn, vg2 + (size_t)l * DIM, vbe2 + (size_t)l * DIM,
               G_GRAPH, DIM, 1);
        }
    }

    // hg = segment_sum(h, batch)
    zero(hg, (size_t)G_GRAPH * DIM);
    k_seg_sum<<<gridN, bk96, 0, stream>>>(h, batch, hg);

    // th = gelu( [hg | fp] @ pW1 + pb1 )  — split-K over the concat
    gemm(hg, DIM, pW1, HEAD_H, pb1, th, HEAD_H, G_GRAPH, DIM, HEAD_H, 0);
    gemm(fp, FP_DIM, pW1 + (size_t)DIM * HEAD_H, HEAD_H, nullptr,
         th, HEAD_H, G_GRAPH, FP_DIM, HEAD_H, 1);
    {
        size_t n4 = (size_t)G_GRAPH * HEAD_H / 4;
        k_gelu4<<<dim3((unsigned)cdiv(n4, 256)), dim3(256), 0, stream>>>(
            (float4*)th, n4);
    }

    // out = th @ pW2 + pb2
    gemm(th, HEAD_H, pW2, NUM_TASK, pb2, out, NUM_TASK,
         G_GRAPH, HEAD_H, NUM_TASK, 0);
}
